// Qa_attention_Emb_14285061226612
// MI455X (gfx1250) — matrix operand order repaired and checked by reading, not compiled, not run on hardware
//
#include <hip/hip_runtime.h>
#include <cstdint>

// ---------------------------------------------------------------------------
// Problem constants (from reference setup_inputs)
// ---------------------------------------------------------------------------
static constexpr int B_  = 32;
static constexpr int La_ = 512;
static constexpr int Lq_ = 64;
static constexpr int H_  = 1024;

// ---------------------------------------------------------------------------
// CDNA5 WMMA types
// ---------------------------------------------------------------------------
typedef __attribute__((ext_vector_type(16))) __bf16   v16bf;
typedef __attribute__((ext_vector_type(8)))  float    v8f;
typedef __attribute__((ext_vector_type(8)))  uint32_t v8u;

union FragU {
    v8u   u;
    v16bf b;
};

__device__ __forceinline__ uint16_t f2bf(float f) {
    union { float f; uint32_t u; } v;
    v.f = f;
    uint32_t r = v.u + 0x7fffu + ((v.u >> 16) & 1u);   // round-to-nearest-even
    return (uint16_t)(r >> 16);
}

// ---------------------------------------------------------------------------
// Async global -> LDS copy (CDNA5 GLOBAL_LOAD_ASYNC_TO_LDS_B128, ASYNCcnt),
// with a synchronous load+ds_store fallback if the builtin is absent.
// Builtin signature (from compiler diagnostic): param0 = int4-vector pointer
// in global (AS1), param1 = LDS (AS3), then (imm offset, imm cpol).
// ---------------------------------------------------------------------------
#if defined(__has_builtin)
#if __has_builtin(__builtin_amdgcn_global_load_async_to_lds_b128) && \
    __has_builtin(__builtin_amdgcn_s_wait_asynccnt)
#define USE_ASYNC_LDS 1
#endif
#endif

typedef int v4i_gcc __attribute__((vector_size(16)));

__device__ __forceinline__ void copy16(const uint16_t* g, uint16_t* l) {
#if USE_ASYNC_LDS
    __builtin_amdgcn_global_load_async_to_lds_b128(
        (__attribute__((address_space(1))) v4i_gcc*)(uint16_t*)g,
        (__attribute__((address_space(3))) v4i_gcc*)l, 0, 0);
#else
    *(uint4*)l = *(const uint4*)g;
#endif
}

template <int N>
__device__ __forceinline__ void wait_copies() {
#if USE_ASYNC_LDS
    __builtin_amdgcn_s_wait_asynccnt(N);
#endif
}

// ---------------------------------------------------------------------------
// f32 -> bf16 conversion (grid-stride)
// ---------------------------------------------------------------------------
__global__ void cvt_f32_bf16(const float* __restrict__ in,
                             uint16_t* __restrict__ out, long long n) {
    long long i = (long long)blockIdx.x * blockDim.x + threadIdx.x;
    long long stride = (long long)gridDim.x * blockDim.x;
    for (; i < n; i += stride) out[i] = f2bf(in[i]);
}

// ---------------------------------------------------------------------------
// Fused f32 -> bf16 convert + transpose through LDS (32x32 tiles, padded).
// in: [R,C] f32 (per batch), out: [C,R] bf16. grid=(C/32,R/32,batch), blk=(32,8)
// ---------------------------------------------------------------------------
__global__ __launch_bounds__(256) void transpose_cvt(
    const float* __restrict__ in, uint16_t* __restrict__ out,
    int R, int C, long long sIn, long long sOut) {
    __shared__ uint16_t tile[32][33];
    const float* ib = in  + (long long)blockIdx.z * sIn;
    uint16_t*    ob = out + (long long)blockIdx.z * sOut;
    int x  = blockIdx.x * 32 + threadIdx.x;
    int y0 = blockIdx.y * 32;
#pragma unroll
    for (int j = threadIdx.y; j < 32; j += 8)
        tile[j][threadIdx.x] = f2bf(ib[(long long)(y0 + j) * C + x]);
    __syncthreads();
    int xo  = y0 + threadIdx.x;
    int yo0 = blockIdx.x * 32;
#pragma unroll
    for (int j = threadIdx.y; j < 32; j += 8)
        ob[(long long)(yo0 + j) * R + xo] = tile[threadIdx.x][j];
}

// ---------------------------------------------------------------------------
// Shared epilogue for the WMMA GEMMs.
//   EPI 0: store f32 C     EPI 1: store bf16 C
//   EPI 2: relu + atomic max-reduce over rows into Cout[b*sC + n]
// C/D layout: VGPR v -> row (tile + v + 8*half), col (tile + r)
// ---------------------------------------------------------------------------
template <int EPI>
__device__ __forceinline__ void epilogue(
    v8f (&accs)[2][2], void* Cout, int b, long long sC, int ldc,
    int m0, int n0, int half, int r) {
    if (EPI == 0) {
        float* C = (float*)Cout + (long long)b * sC;
#pragma unroll
        for (int mi = 0; mi < 2; ++mi)
#pragma unroll
            for (int ni = 0; ni < 2; ++ni)
#pragma unroll
                for (int v = 0; v < 8; ++v)
                    C[(long long)(m0 + mi * 16 + v + 8 * half) * ldc +
                      n0 + ni * 16 + r] = accs[mi][ni][v];
    } else if (EPI == 1) {
        uint16_t* C = (uint16_t*)Cout + (long long)b * sC;
#pragma unroll
        for (int mi = 0; mi < 2; ++mi)
#pragma unroll
            for (int ni = 0; ni < 2; ++ni)
#pragma unroll
                for (int v = 0; v < 8; ++v)
                    C[(long long)(m0 + mi * 16 + v + 8 * half) * ldc +
                      n0 + ni * 16 + r] = f2bf(accs[mi][ni][v]);
    } else {
        float* red = (float*)Cout + (long long)b * sC;
#pragma unroll
        for (int ni = 0; ni < 2; ++ni) {
            float x = 0.0f;   // relu folded into max (buffer zero-initialized)
#pragma unroll
            for (int mi = 0; mi < 2; ++mi)
#pragma unroll
                for (int v = 0; v < 8; ++v)
                    x = fmaxf(x, accs[mi][ni][v]);
            atomicMax((unsigned int*)(red + n0 + ni * 16 + r),
                      __float_as_uint(x));
        }
    }
}

// ---------------------------------------------------------------------------
// LDS-staged, double-buffered NT GEMM (C = A * B^T, A:[M,K], B:[N,K] bf16).
// 4 waves / 128 threads per block compute a 64x64 C tile; each k-step the
// block async-DMAs the 64x32 A and B panels into LDS (ASYNCcnt), overlapped
// with wmmas on the previous panel. Requires M,N % 64 == 0 and K % 32 == 0.
// ---------------------------------------------------------------------------
template <int EPI>
__global__ __launch_bounds__(128) void gemm_bf16_wmma_lds(
    const uint16_t* __restrict__ A, const uint16_t* __restrict__ Bm,
    void* __restrict__ Cout,
    int M, int N, int K,
    long long sA, long long sB, long long sC,
    int lda, int ldb, int ldc) {

    // [buf][A|B][row][32 data + 8 pad]  (row stride 80B: 16B-aligned, 20 banks)
    __shared__ uint16_t sm[2][2][64][40];

    const int t    = threadIdx.x;
    const int lane = t & 31;
    const int wave = t >> 5;
    const int half = lane >> 4;
    const int r    = lane & 15;
    const int mblk = blockIdx.y * 64;
    const int nblk = blockIdx.x * 64;
    const int b    = blockIdx.z;

    const uint16_t* Ab = A  + (long long)b * sA;
    const uint16_t* Bb = Bm + (long long)b * sB;

    // each thread copies 2 chunks (16B) per panel, 256 chunks per 4KB panel
    const int idx0 = t * 2, idx1 = t * 2 + 1;
    const int row0 = idx0 >> 2, ch0 = (idx0 & 3) * 8;
    const int row1 = idx1 >> 2, ch1 = (idx1 & 3) * 8;

    auto issue = [&](int buf, int k0) {
        copy16(Ab + (long long)(mblk + row0) * lda + k0 + ch0, &sm[buf][0][row0][ch0]);
        copy16(Ab + (long long)(mblk + row1) * lda + k0 + ch1, &sm[buf][0][row1][ch1]);
        copy16(Bb + (long long)(nblk + row0) * ldb + k0 + ch0, &sm[buf][1][row0][ch0]);
        copy16(Bb + (long long)(nblk + row1) * ldb + k0 + ch1, &sm[buf][1][row1][ch1]);
    };

    v8f accs[2][2] = {};
    const int mloc0 = (wave >> 1) * 32 + r;
    const int nloc0 = (wave & 1) * 32 + r;

    issue(0, 0);
    const int KT = K >> 5;
    for (int kt = 0; kt < KT; ++kt) {
        const int buf = kt & 1;
        if (kt + 1 < KT) {
            issue(buf ^ 1, (kt + 1) << 5);
            wait_copies<4>();   // previous group of 4 async ops done (in-order)
        } else {
            wait_copies<0>();
        }
        __syncthreads();        // all waves' panel copies visible

        FragU fa0, fa1, fb0, fb1;
        const uint16_t* ra0 = &sm[buf][0][mloc0][0];
        const uint16_t* ra1 = &sm[buf][0][mloc0 + 16][0];
        const uint16_t* rb0 = &sm[buf][1][nloc0][0];
        const uint16_t* rb1 = &sm[buf][1][nloc0 + 16][0];
        // A frag chunks (elements): 8*half and 16 + 8*half
        {
            uint4 x0 = *(const uint4*)(ra0 + 8 * half);
            uint4 x1 = *(const uint4*)(ra0 + 16 + 8 * half);
            fa0.u = (v8u){x0.x, x0.y, x0.z, x0.w, x1.x, x1.y, x1.z, x1.w};
            uint4 y0 = *(const uint4*)(ra1 + 8 * half);
            uint4 y1 = *(const uint4*)(ra1 + 16 + 8 * half);
            fa1.u = (v8u){y0.x, y0.y, y0.z, y0.w, y1.x, y1.y, y1.z, y1.w};
        }
        // B frag chunks (elements): 8*half and 16 + 8*half
        {
            uint4 x0 = *(const uint4*)(rb0 + 8 * half);
            uint4 x1 = *(const uint4*)(rb0 + 16 + 8 * half);
            fb0.u = (v8u){x0.x, x0.y, x0.z, x0.w, x1.x, x1.y, x1.z, x1.w};
            uint4 y0 = *(const uint4*)(rb1 + 8 * half);
            uint4 y1 = *(const uint4*)(rb1 + 16 + 8 * half);
            fb1.u = (v8u){y0.x, y0.y, y0.z, y0.w, y1.x, y1.y, y1.z, y1.w};
        }
        accs[0][0] = __builtin_amdgcn_wmma_f32_16x16x32_bf16(false, fa0.b, false, fb0.b, (short)0, accs[0][0], false, false);
        accs[0][1] = __builtin_amdgcn_wmma_f32_16x16x32_bf16(false, fa0.b, false, fb1.b, (short)0, accs[0][1], false, false);
        accs[1][0] = __builtin_amdgcn_wmma_f32_16x16x32_bf16(false, fa1.b, false, fb0.b, (short)0, accs[1][0], false, false);
        accs[1][1] = __builtin_amdgcn_wmma_f32_16x16x32_bf16(false, fa1.b, false, fb1.b, (short)0, accs[1][1], false, false);
        __syncthreads();        // reads done before this buffer is re-filled
    }

    epilogue<EPI>(accs, Cout, b, sC, ldc,
                  mblk + (wave >> 1) * 32, nblk + (wave & 1) * 32, half, r);
}

// ---------------------------------------------------------------------------
// Direct-load batched GEMM (for the small / short-K GEMMs), 2x2 blocked.
//   BNT == true : B given [N,K] (NT, b128 loads); false: [K,N] (NN, strided)
// ---------------------------------------------------------------------------
template <bool BNT, int EPI>
__global__ __launch_bounds__(128) void gemm_bf16_wmma(
    const uint16_t* __restrict__ A, const uint16_t* __restrict__ Bm,
    void* __restrict__ Cout,
    int M, int N, int K,
    long long sA, long long sB, long long sC,
    int lda, int ldb, int ldc) {

    const int lane = threadIdx.x & 31;
    const int wave = threadIdx.x >> 5;
    const int half = lane >> 4;
    const int r    = lane & 15;
    const int n0   = blockIdx.x * 64 + (wave & 1) * 32;
    const int m0   = blockIdx.y * 64 + (wave >> 1) * 32;
    const int b    = blockIdx.z;
    if (m0 >= M || n0 >= N) return;

    const uint16_t* Ab = A  + (long long)b * sA;
    const uint16_t* Bb = Bm + (long long)b * sB;

    v8f accs[2][2] = {};
    const uint16_t* arow0 = Ab + (long long)(m0 + r) * lda;
    const uint16_t* arow1 = Ab + (long long)(m0 + 16 + r) * lda;
    const uint16_t* brow0 = Bb + (long long)(n0 + r) * ldb;
    const uint16_t* brow1 = Bb + (long long)(n0 + 16 + r) * ldb;
    const uint16_t* bcol0 = Bb + (n0 + r);
    const uint16_t* bcol1 = Bb + (n0 + 16 + r);

    for (int k0 = 0; k0 < K; k0 += 32) {
        if (k0 + 128 < K) {
            __builtin_prefetch(arow0 + k0 + 128, 0, 3);
            __builtin_prefetch(arow1 + k0 + 128, 0, 3);
            if (BNT) {
                __builtin_prefetch(brow0 + k0 + 128, 0, 3);
                __builtin_prefetch(brow1 + k0 + 128, 0, 3);
            }
        }
        FragU fa0, fa1, fb0, fb1;
#pragma unroll
        for (int v = 0; v < 8; ++v) {
            int ka = k0 + (half << 4) + (v << 1);
            fa0.u[v] = *(const uint32_t*)(arow0 + ka);
            fa1.u[v] = *(const uint32_t*)(arow1 + ka);
        }
        if (BNT) {
#pragma unroll
            for (int v = 0; v < 8; ++v) {
                int kb = k0 + (half << 4) + (v << 1);
                fb0.u[v] = *(const uint32_t*)(brow0 + kb);
                fb1.u[v] = *(const uint32_t*)(brow1 + kb);
            }
        } else {
#pragma unroll
            for (int v = 0; v < 8; ++v) {
                int kb = k0 + (half << 4) + (v << 1);
                uint32_t lo0 = bcol0[(long long)kb * ldb];
                uint32_t hi0 = bcol0[(long long)(kb + 1) * ldb];
                uint32_t lo1 = bcol1[(long long)kb * ldb];
                uint32_t hi1 = bcol1[(long long)(kb + 1) * ldb];
                fb0.u[v] = lo0 | (hi0 << 16);
                fb1.u[v] = lo1 | (hi1 << 16);
            }
        }
        accs[0][0] = __builtin_amdgcn_wmma_f32_16x16x32_bf16(false, fa0.b, false, fb0.b, (short)0, accs[0][0], false, false);
        accs[0][1] = __builtin_amdgcn_wmma_f32_16x16x32_bf16(false, fa0.b, false, fb1.b, (short)0, accs[0][1], false, false);
        accs[1][0] = __builtin_amdgcn_wmma_f32_16x16x32_bf16(false, fa1.b, false, fb0.b, (short)0, accs[1][0], false, false);
        accs[1][1] = __builtin_amdgcn_wmma_f32_16x16x32_bf16(false, fa1.b, false, fb1.b, (short)0, accs[1][1], false, false);
    }
    epilogue<EPI>(accs, Cout, b, sC, ldc, m0, n0, half, r);
}

// ---------------------------------------------------------------------------
// Row softmax (f32 in, bf16 out). One wave per row, blockDim = (32, 8).
// ---------------------------------------------------------------------------
__global__ void softmax_rows(const float* __restrict__ S,
                             uint16_t* __restrict__ P, int R, int C) {
    int row = blockIdx.x * blockDim.y + threadIdx.y;
    if (row >= R) return;
    const float* s = S + (long long)row * C;
    float m = -3.4e38f;
    for (int c = threadIdx.x; c < C; c += 32) m = fmaxf(m, s[c]);
#pragma unroll
    for (int i = 16; i > 0; i >>= 1) m = fmaxf(m, __shfl_xor(m, i, 32));
    float sum = 0.0f;
    for (int c = threadIdx.x; c < C; c += 32) sum += __expf(s[c] - m);
#pragma unroll
    for (int i = 16; i > 0; i >>= 1) sum += __shfl_xor(sum, i, 32);
    float inv = __frcp_rn(sum);
    uint16_t* p = P + (long long)row * C;
    for (int c = threadIdx.x; c < C; c += 32) p[c] = f2bf(__expf(s[c] - m) * inv);
}

// ---------------------------------------------------------------------------
// Final combine: out = 0.5 * S_aq + 0.5 * S_qa
// ---------------------------------------------------------------------------
__global__ void combine_out(const float* __restrict__ sa,
                            const float* __restrict__ sq,
                            float* __restrict__ out, int n) {
    int i = blockIdx.x * blockDim.x + threadIdx.x;
    if (i < n) out[i] = 0.5f * sa[i] + 0.5f * sq[i];
}

// ---------------------------------------------------------------------------
// Host launcher
// ---------------------------------------------------------------------------
static inline uint8_t* carve(uint8_t*& p, size_t bytes) {
    uint8_t* q = p;
    p += (bytes + 255) & ~size_t(255);
    return q;
}

extern "C" void kernel_launch(void* const* d_in, const int* in_sizes, int n_in,
                              void* d_out, int out_size, void* d_ws, size_t ws_size,
                              hipStream_t stream) {
    (void)in_sizes; (void)n_in; (void)out_size; (void)ws_size;

    const float* A_f   = (const float*)d_in[0];   // [B, La, H]
    const float* Q_f   = (const float*)d_in[1];   // [B, Lq, H]
    const float* W11_f = (const float*)d_in[2];   // [H, H]
    const float* W12_f = (const float*)d_in[3];
    const float* W21_f = (const float*)d_in[4];
    const float* W22_f = (const float*)d_in[5];
    float* out = (float*)d_out;                   // [B, H]

    const long long nA = (long long)B_ * La_ * H_;
    const long long nQ = (long long)B_ * Lq_ * H_;
    const long long nW = (long long)H_ * H_;

    uint8_t* p = (uint8_t*)d_ws;
    uint16_t* A_bf  = (uint16_t*)carve(p, nA * 2);
    uint16_t* Q_bf  = (uint16_t*)carve(p, nQ * 2);
    uint16_t* Qt_bf = (uint16_t*)carve(p, nQ * 2);   // [B, H, Lq]
    uint16_t* W11_b = (uint16_t*)carve(p, nW * 2);
    uint16_t* W12t  = (uint16_t*)carve(p, nW * 2);
    uint16_t* W21t  = (uint16_t*)carve(p, nW * 2);
    uint16_t* W22t  = (uint16_t*)carve(p, nW * 2);
    uint16_t* P1    = (uint16_t*)carve(p, nQ * 2);
    uint16_t* P2    = (uint16_t*)carve(p, nQ * 2);
    float*    aq_s  = (float*)   carve(p, (long long)B_*La_*Lq_ * 4);
    float*    qa_s  = (float*)   carve(p, (long long)B_*Lq_*La_ * 4);
    uint16_t* aqP   = (uint16_t*)carve(p, (long long)B_*La_*Lq_ * 2);
    uint16_t* qaP   = (uint16_t*)carve(p, (long long)B_*Lq_*La_ * 2);
    uint16_t* Xa    = (uint16_t*)carve(p, nA * 2);
    uint16_t* Xq    = (uint16_t*)carve(p, nQ * 2);
    float*    Smax_a= (float*)   carve(p, (long long)B_*H_ * 4);
    float*    Smax_q= (float*)   carve(p, (long long)B_*H_ * 4);

    // --- f32 -> bf16 (+ transposed copies so hot GEMMs are all NT) ---
    cvt_f32_bf16<<<4096, 256, 0, stream>>>(A_f,   A_bf,  nA);
    cvt_f32_bf16<<<2048, 256, 0, stream>>>(Q_f,   Q_bf,  nQ);
    cvt_f32_bf16<<<1024, 256, 0, stream>>>(W11_f, W11_b, nW);
    transpose_cvt<<<dim3(H_/32, H_/32, 1),  dim3(32, 8), 0, stream>>>(
        W12_f, W12t, H_, H_, 0, 0);
    transpose_cvt<<<dim3(H_/32, H_/32, 1),  dim3(32, 8), 0, stream>>>(
        W21_f, W21t, H_, H_, 0, 0);
    transpose_cvt<<<dim3(H_/32, H_/32, 1),  dim3(32, 8), 0, stream>>>(
        W22_f, W22t, H_, H_, 0, 0);
    transpose_cvt<<<dim3(H_/32, Lq_/32, B_), dim3(32, 8), 0, stream>>>(
        Q_f, Qt_bf, Lq_, H_, (long long)Lq_*H_, (long long)H_*Lq_);

    (void)hipMemsetAsync(Smax_a, 0, (size_t)B_ * H_ * 4, stream);
    (void)hipMemsetAsync(Smax_q, 0, (size_t)B_ * H_ * 4, stream);

    const long long sQ  = (long long)Lq_ * H_;
    const long long sAa = (long long)La_ * H_;

    // P1[b] = Q_b @ W11^T                                          [B,Lq,H]
    gemm_bf16_wmma_lds<1><<<dim3(H_/64, 1, B_), 128, 0, stream>>>(
        Q_bf, W11_b, P1, Lq_, H_, H_, sQ, 0, sQ, H_, H_, H_);
    // P2[b] = Q_b @ W12  (via W12^T)                               [B,Lq,H]
    gemm_bf16_wmma_lds<1><<<dim3(H_/64, 1, B_), 128, 0, stream>>>(
        Q_bf, W12t, P2, Lq_, H_, H_, sQ, 0, sQ, H_, H_, H_);
    // aq_scores[b] = A_b @ P1_b^T                                  [B,La,Lq]
    gemm_bf16_wmma_lds<0><<<dim3(Lq_/64, La_/64, B_), 128, 0, stream>>>(
        A_bf, P1, aq_s, La_, Lq_, H_, sAa, sQ, (long long)La_*Lq_, H_, H_, Lq_);
    // qa_scores[b] = P2_b @ A_b^T                                  [B,Lq,La]
    gemm_bf16_wmma_lds<0><<<dim3(La_/64, 1, B_), 128, 0, stream>>>(
        P2, A_bf, qa_s, Lq_, La_, H_, sQ, sAa, (long long)Lq_*La_, H_, H_, La_);

    // softmaxes
    softmax_rows<<<(B_*La_)/8, dim3(32, 8), 0, stream>>>(aq_s, aqP, B_*La_, Lq_);
    softmax_rows<<<(B_*Lq_)/8, dim3(32, 8), 0, stream>>>(qa_s, qaP, B_*Lq_, La_);

    // X_a[b] = aq_att_b @ Q_b  (K=64, direct NT path)              [B,La,H]
    gemm_bf16_wmma<true, 1><<<dim3(H_/64, La_/64, B_), 128, 0, stream>>>(
        aqP, Qt_bf, Xa, La_, H_, Lq_, (long long)La_*Lq_, sQ, sAa, Lq_, Lq_, H_);
    // X_q[b] = qa_att_b @ A_b  (small; NN strided path)            [B,Lq,H]
    gemm_bf16_wmma<false, 1><<<dim3(H_/64, 1, B_), 128, 0, stream>>>(
        qaP, A_bf, Xq, Lq_, H_, La_, (long long)Lq_*La_, sAa, sQ, La_, H_, H_);

    // S_aq[b,:] = max_a relu(X_a @ W21)  (dominant GEMM, LDS pipeline)
    gemm_bf16_wmma_lds<2><<<dim3(H_/64, La_/64, B_), 128, 0, stream>>>(
        Xa, W21t, Smax_a, La_, H_, H_, sAa, 0, (long long)H_, H_, H_, H_);
    // S_qa[b,:] = max_q relu(X_q @ W22)
    gemm_bf16_wmma_lds<2><<<dim3(H_/64, 1, B_), 128, 0, stream>>>(
        Xq, W22t, Smax_q, Lq_, H_, H_, sQ, 0, (long long)H_, H_, H_, H_);

    // out = 0.5*S_aq + 0.5*S_qa
    combine_out<<<(B_*H_ + 255)/256, 256, 0, stream>>>(Smax_a, Smax_q, out, B_*H_);
}
